// FLAModel_14379550507697
// MI455X (gfx1250) — compile-verified, matrix operand order checked by reading, not run on hardware
//
#include <hip/hip_runtime.h>
#include <math.h>

// ---------------- model constants (match reference) ----------------
#define TKN   4096      // B*S tokens
#define SEQ   2048
#define BATCH 2
#define DIM   512
#define DKc   256
#define DVc   512
#define NH    8
#define HKc   32
#define HVc   64
#define VOC   50257
#define NL    4
#define FF    2048
#define EPSc  1e-5f

typedef __attribute__((ext_vector_type(16))) __bf16          v16bf;
typedef __attribute__((ext_vector_type(8)))  float           v8f;
typedef __attribute__((ext_vector_type(4)))  unsigned int    v4u;
typedef __attribute__((ext_vector_type(4)))  float           v4f;
typedef __attribute__((ext_vector_type(4)))  unsigned short  v4us;

// ---------------- gfx1250 async-to-LDS support (guarded) ----------------
#if defined(__has_builtin)
#  if __has_builtin(__builtin_amdgcn_global_load_async_to_lds_b128) && \
      __has_builtin(__builtin_amdgcn_s_wait_asynccnt)
#    define USE_ASYNC_LDS 1
#  endif
#endif
#ifndef USE_ASYNC_LDS
#  define USE_ASYNC_LDS 0
#endif

#if USE_ASYNC_LDS
// builtin signature (from compiler diagnostic): (<4 x i32> __device__*, <4 x i32> __shared__*, imm, imm)
typedef int v4i_g __attribute__((vector_size(16)));
typedef __attribute__((address_space(1))) v4i_g gv4i;   // global ("__device__")
typedef __attribute__((address_space(3))) v4i_g lv4i;   // LDS ("__shared__")
// 16-byte async DMA: global -> LDS, tracked on ASYNCcnt (no VGPR data path)
#define ASYNC_CP16(ldst, gsrc)                                                \
    __builtin_amdgcn_global_load_async_to_lds_b128(                           \
        (gv4i*)(const void*)(gsrc), (lv4i*)(void*)(ldst), 0, 0)
#define ASYNC_WAIT() __builtin_amdgcn_s_wait_asynccnt(0)
#endif

// ---------------- helpers ----------------
__device__ __forceinline__ unsigned short f2bf(float f) {
    union { float f; unsigned u; } x; x.f = f;
    unsigned u = x.u + 0x7FFFu + ((x.u >> 16) & 1u);   // round-to-nearest-even
    return (unsigned short)(u >> 16);
}

__device__ __forceinline__ float gelu_exact(float x) {
    return 0.5f * x * (1.0f + erff(x * 0.70710678118654752f));
}

__device__ __forceinline__ float logsig_div16(float x) {
    float m = fminf(x, 0.0f);
    return (m - log1pf(expf(-fabsf(x)))) * 0.0625f;     // log_sigmoid(x)/16
}

// bf16 WMMA fragment for one lane: two contiguous 16B chunks of the K-major row.
// ISA 16-bit A/B 16x32 layout: lanes 0-15 hold K 0-7 & 16-23, lanes 16-31 hold
// K 8-15 & 24-31 (khalf = 0 or 8); vector elems 0-7 = chunk0, elems 8-15 = chunk1.
__device__ __forceinline__ v16bf load_frag(const unsigned short* rowp, int off) {
    union { v16bf v; v4u u[2]; } f;
    f.u[0] = *(const v4u*)(rowp + off);        // K = off .. off+7
    f.u[1] = *(const v4u*)(rowp + off + 16);   // K = off+16 .. off+23
    return f.v;
}

// ---------------- GEMM: C[M,N] = A[M,K] @ W[N,K]^T  (bf16 in, fp32 acc) ----------
// epi: 0=none 1=+bias 2=+bias,gelu 3=+res 4=+bias,+res 5=+bias,log_sigmoid/16
// Cf (fp32) and/or Cb (bf16) outputs, leading dim ldc. K must be a multiple of 32.
// Block = 128 threads (4 waves); wave w owns tile [by*64 + w*16, bx*64 .. +64).
__global__ __launch_bounds__(128) void gemm_bf16_wmma(
    const unsigned short* __restrict__ A, const unsigned short* __restrict__ W,
    float* __restrict__ Cf, unsigned short* __restrict__ Cb, int ldc,
    int M, int N, int K,
    const float* __restrict__ bias, const float* __restrict__ res, int epi)
{
    const int lane   = threadIdx.x & 31;
    const int wave   = threadIdx.x >> 5;
    const int tileM  = blockIdx.y * 64 + wave * 16;
    const int tileN  = blockIdx.x * 64;
    const int lane15 = lane & 15;
    const int khalf  = (lane & 16) ? 8 : 0;

    const unsigned short* arow = A + (size_t)(tileM + lane15) * K;
    const unsigned short* brow[4];
#pragma unroll
    for (int t = 0; t < 4; ++t) {
        int col = tileN + t * 16 + lane15;
        brow[t] = W + (size_t)(col < N ? col : 0) * K;  // clamped; OOB cols never stored
    }

    v8f z = {0.f, 0.f, 0.f, 0.f, 0.f, 0.f, 0.f, 0.f};
    v8f cacc[4] = {z, z, z, z};

    for (int k0 = 0; k0 < K; k0 += 32) {
        const int off = k0 + khalf;
        __builtin_prefetch(arow + k0 + 256, 0, 1);      // global_prefetch_b8
        v16bf a = load_frag(arow, off);
        v16bf b[4];
#pragma unroll
        for (int t = 0; t < 4; ++t) b[t] = load_frag(brow[t], off);
#pragma unroll
        for (int t = 0; t < 4; ++t)
            cacc[t] = __builtin_amdgcn_wmma_f32_16x16x32_bf16(
                false, a, false, b[t], (short)0, cacc[t], false, false);
    }

    const int rbase = tileM + ((lane & 16) ? 8 : 0);    // C layout: reg r -> row (+8 hi lanes)
#pragma unroll
    for (int t = 0; t < 4; ++t) {
        int col = tileN + t * 16 + lane15;
        if (col >= N) continue;
        float bv = (epi == 1 || epi == 2 || epi == 4 || epi == 5) ? bias[col] : 0.0f;
#pragma unroll
        for (int r = 0; r < 8; ++r) {
            int row = rbase + r;
            float acc = cacc[t][r] + bv;
            if (epi == 2) acc = gelu_exact(acc);
            if (epi == 5) acc = logsig_div16(acc);
            if (epi == 3 || epi == 4) acc += res[(size_t)row * ldc + col];
            size_t idx = (size_t)row * ldc + col;
            if (Cf) Cf[idx] = acc;
            if (Cb) Cb[idx] = f2bf(acc);
        }
    }
}

// ---------------- fp32 -> bf16 conversion (4 elems/thread) ----------------
__global__ void cvt_bf16_kernel(const float* __restrict__ in,
                                unsigned short* __restrict__ out, int n4)
{
    int i = blockIdx.x * 256 + threadIdx.x;
    if (i >= n4) return;
    v4f f = ((const v4f*)in)[i];
    v4us r = { f2bf(f[0]), f2bf(f[1]), f2bf(f[2]), f2bf(f[3]) };
    ((v4us*)out)[i] = r;
}

// wgk2 [NL,256,16] fp32 -> [NL,256,32] bf16, zero-padded cols 16..31
__global__ void cvt_pad_wgk2_kernel(const float* __restrict__ in,
                                    unsigned short* __restrict__ out)
{
    int i = blockIdx.x * 256 + threadIdx.x;      // over NL*DKc*32
    if (i >= NL * DKc * 32) return;
    int c = i & 31, r = i >> 5;
    out[i] = (c < 16) ? f2bf(in[r * 16 + c]) : (unsigned short)0;
}

__global__ void fill0_bf16_kernel(unsigned short* __restrict__ p, int n4)
{
    int i = blockIdx.x * 256 + threadIdx.x;
    if (i >= n4) return;
    v4us zz = { 0, 0, 0, 0 };
    ((v4us*)p)[i] = zz;
}

// ---------------- embedding gather ----------------
__global__ void embed_kernel(const int* __restrict__ tok, const float* __restrict__ emb,
                             float* __restrict__ x)
{
    int t = blockIdx.x;
    const float* src = emb + (size_t)tok[t] * DIM;
    for (int d = threadIdx.x; d < DIM; d += blockDim.x)
        x[(size_t)t * DIM + d] = src[d];
}

// ---------------- LayerNorm -> bf16 (one token per block, 256 threads, D=512) ----
__global__ __launch_bounds__(256) void ln_kernel(const float* __restrict__ in,
                                                 const float* __restrict__ w,
                                                 const float* __restrict__ b,
                                                 unsigned short* __restrict__ out)
{
    __shared__ float s1[256], s2[256];
    int t = blockIdx.x, tid = threadIdx.x;
    float a0 = in[(size_t)t * DIM + tid];
    float a1 = in[(size_t)t * DIM + tid + 256];
    s1[tid] = a0 + a1;
    s2[tid] = a0 * a0 + a1 * a1;
    __syncthreads();
    for (int s = 128; s > 0; s >>= 1) {
        if (tid < s) { s1[tid] += s1[tid + s]; s2[tid] += s2[tid + s]; }
        __syncthreads();
    }
    float mean = s1[0] * (1.0f / DIM);
    float var  = s2[0] * (1.0f / DIM) - mean * mean;
    float inv  = rsqrtf(var + EPSc);
    out[(size_t)t * DIM + tid]       = f2bf((a0 - mean) * inv * w[tid]       + b[tid]);
    out[(size_t)t * DIM + tid + 256] = f2bf((a1 - mean) * inv * w[tid + 256] + b[tid + 256]);
}

// ---------------- GLA recurrence ----------------
// grid (NH, BATCH), block 128. Lane pair (2j,2j+1) owns state column j: each of
// the two lanes carries 16 of the 32 K-rows in VGPRs; partial outputs combined
// with an intra-wave shfl_xor. Chunked staging of raw q,k,v via async DMA to LDS
// (ASYNCcnt) when available; exp(gk) staged through VALU. q-scale folded into the
// output store (o = scale * sum q_i S_i), so q can be DMA'd untouched.
#define CHg 16
__global__ __launch_bounds__(128) void gla_kernel(const float* __restrict__ q,
                                                  const float* __restrict__ k,
                                                  const float* __restrict__ v,
                                                  const float* __restrict__ gk,
                                                  float* __restrict__ o)
{
    __shared__ float qs[CHg][HKc], ks[CHg][HKc], gs[CHg][HKc], vs[CHg][HVc];
    const int head = blockIdx.x, batch = blockIdx.y, tid = threadIdx.x;
    const int j  = tid >> 1;          // value column 0..63
    const int ih = tid & 1;           // which half of the K rows
    const int tok0 = batch * SEQ;
    const float scale = 0.17677669529663687f;  // HK^-0.5

    float S[16];
#pragma unroll
    for (int i = 0; i < 16; ++i) S[i] = 0.0f;

    for (int c = 0; c < SEQ; c += CHg) {
        __syncthreads();              // previous chunk fully consumed before refill
#if USE_ASYNC_LDS
        {
            // qs/ks: CHg*HKc = 512 floats = 128 x 16B -> one DMA per thread each
            int st = tid >> 3, f4 = (tid & 7) * 4;
            size_t gbase = (size_t)(tok0 + c + st) * DKc + head * HKc + f4;
            ASYNC_CP16(&qs[st][f4], q + gbase);
            ASYNC_CP16(&ks[st][f4], k + gbase);
            // vs: CHg*HVc = 1024 floats = 256 x 16B -> two DMAs per thread
#pragma unroll
            for (int t2 = 0; t2 < 2; ++t2) {
                int i2 = tid + t2 * 128;
                int stv = i2 >> 4, fv = (i2 & 15) * 4;
                ASYNC_CP16(&vs[stv][fv],
                           v + (size_t)(tok0 + c + stv) * DVc + head * HVc + fv);
            }
        }
        for (int idx = tid; idx < CHg * HKc; idx += 128) {
            int st = idx >> 5, i = idx & 31;
            gs[st][i] = expf(gk[(size_t)(tok0 + c + st) * DKc + head * HKc + i]);
        }
        ASYNC_WAIT();                 // s_wait_asynccnt 0
        __syncthreads();
#else
        for (int idx = tid; idx < CHg * HKc; idx += 128) {
            int st = idx >> 5, i = idx & 31;
            size_t base = (size_t)(tok0 + c + st) * DKc + head * HKc + i;
            qs[st][i] = q[base];
            ks[st][i] = k[base];
            gs[st][i] = expf(gk[base]);
        }
        for (int idx = tid; idx < CHg * HVc; idx += 128) {
            int st = idx >> 6, jj = idx & 63;
            vs[st][jj] = v[(size_t)(tok0 + c + st) * DVc + head * HVc + jj];
        }
        __syncthreads();
#endif
        for (int st = 0; st < CHg; ++st) {
            float vj = vs[st][j];
            float acc = 0.0f;
#pragma unroll
            for (int ii = 0; ii < 16; ++ii) {
                int i = ih * 16 + ii;
                S[ii] = S[ii] * gs[st][i] + ks[st][i] * vj;  // diag(exp(gk)) S + k v^T
                acc += qs[st][i] * S[ii];                     // o = q S (scale folded below)
            }
            acc += __shfl_xor(acc, 1, 32);
            if (ih == 0)
                o[(size_t)(tok0 + c + st) * DVc + head * HVc + j] = acc * scale;
        }
    }
}

// ---------------- fused per-head RMSNorm + swish gate -> bf16 ----------------
// grid (TKN, NH), block 64 (one head-vector per block).
__global__ __launch_bounds__(64) void gatenorm_kernel(const float* __restrict__ o,
                                                      const float* __restrict__ g,
                                                      const float* __restrict__ onw,
                                                      unsigned short* __restrict__ ob)
{
    __shared__ float sm[2];
    int tok = blockIdx.x, head = blockIdx.y, j = threadIdx.x;
    size_t idx = (size_t)tok * DVc + head * HVc + j;
    float val = o[idx];
    float ss = val * val;
#pragma unroll
    for (int d = 16; d > 0; d >>= 1) ss += __shfl_xor(ss, d, 32);
    if ((j & 31) == 0) sm[j >> 5] = ss;
    __syncthreads();
    float inv = rsqrtf((sm[0] + sm[1]) * (1.0f / HVc) + EPSc);
    float gg  = g[idx];
    float sw  = gg / (1.0f + expf(-gg));                 // swish
    ob[idx] = f2bf(val * inv * onw[j] * sw);
}

// ---------------- host orchestration ----------------
static inline void launch_gemm(const unsigned short* A, const unsigned short* W,
                               float* Cf, unsigned short* Cb, int ldc,
                               int M, int N, int K,
                               const float* bias, const float* res, int epi,
                               hipStream_t s)
{
    dim3 grid((N + 63) / 64, M / 64), block(128);
    gemm_bf16_wmma<<<grid, block, 0, s>>>(A, W, Cf, Cb, ldc, M, N, K, bias, res, epi);
}

static inline void launch_cvt(const float* in, unsigned short* out, size_t n, hipStream_t s)
{
    int n4 = (int)(n / 4);
    cvt_bf16_kernel<<<(n4 + 255) / 256, 256, 0, s>>>(in, out, n4);
}

extern "C" void kernel_launch(void* const* d_in, const int* in_sizes, int n_in,
                              void* d_out, int out_size, void* d_ws, size_t ws_size,
                              hipStream_t stream)
{
    (void)in_sizes; (void)n_in; (void)out_size; (void)ws_size;
    const int*   tokens = (const int*)  d_in[0];
    const float* embed  = (const float*)d_in[1];
    const float* ln1_w  = (const float*)d_in[2];
    const float* ln1_b  = (const float*)d_in[3];
    const float* wq     = (const float*)d_in[4];
    const float* wk     = (const float*)d_in[5];
    const float* wv     = (const float*)d_in[6];
    const float* wg     = (const float*)d_in[7];
    const float* wgk1   = (const float*)d_in[8];
    const float* wgk2   = (const float*)d_in[9];
    const float* bgk    = (const float*)d_in[10];
    const float* onw    = (const float*)d_in[11];
    const float* wo     = (const float*)d_in[12];
    const float* ln2_w  = (const float*)d_in[13];
    const float* ln2_b  = (const float*)d_in[14];
    const float* w1     = (const float*)d_in[15];
    const float* b1     = (const float*)d_in[16];
    const float* w2     = (const float*)d_in[17];
    const float* b2     = (const float*)d_in[18];
    const float* lnf_w  = (const float*)d_in[19];
    const float* lnf_b  = (const float*)d_in[20];
    const float* head_w = (const float*)d_in[21];
    const float* head_b = (const float*)d_in[22];
    float* out = (float*)d_out;

    // -------- workspace layout (16B-aligned carve-out, ~148 MB) --------
    char* base = (char*)d_ws;
    size_t off = 0;
    auto allocf = [&](size_t n) { float* p = (float*)(base + off);
                                  off += ((n * 4 + 15) & ~(size_t)15); return p; };
    auto allocb = [&](size_t n) { unsigned short* p = (unsigned short*)(base + off);
                                  off += ((n * 2 + 15) & ~(size_t)15); return p; };
    // fp32 activations
    float* x  = allocf((size_t)TKN * DIM);
    float* q  = allocf((size_t)TKN * DKc);
    float* k  = allocf((size_t)TKN * DKc);
    float* v  = allocf((size_t)TKN * DVc);
    float* g  = allocf((size_t)TKN * DVc);
    float* gk = allocf((size_t)TKN * DKc);
    float* o  = allocf((size_t)TKN * DVc);
    // bf16 activations
    unsigned short* h    = allocb((size_t)TKN * DIM);
    unsigned short* gtb  = allocb((size_t)TKN * 32);      // [T,32], cols 16.. zeroed
    unsigned short* midb = allocb((size_t)TKN * FF);
    unsigned short* ob   = allocb((size_t)TKN * DVc);
    // bf16 weights
    unsigned short* wqb   = allocb((size_t)NL * DKc * DIM);
    unsigned short* wkb   = allocb((size_t)NL * DKc * DIM);
    unsigned short* wvb   = allocb((size_t)NL * DVc * DIM);
    unsigned short* wgb   = allocb((size_t)NL * DVc * DIM);
    unsigned short* wgk1b = allocb((size_t)NL * 16 * DIM);
    unsigned short* wgk2b = allocb((size_t)NL * DKc * 32);
    unsigned short* wob   = allocb((size_t)NL * DIM * DVc);
    unsigned short* w1b   = allocb((size_t)NL * FF * DIM);
    unsigned short* w2b   = allocb((size_t)NL * DIM * FF);
    unsigned short* headb = allocb((size_t)VOC * DIM);

    // -------- one-time per-call weight conversion (all L2-resident) --------
    launch_cvt(wq,   wqb,   (size_t)NL * DKc * DIM, stream);
    launch_cvt(wk,   wkb,   (size_t)NL * DKc * DIM, stream);
    launch_cvt(wv,   wvb,   (size_t)NL * DVc * DIM, stream);
    launch_cvt(wg,   wgb,   (size_t)NL * DVc * DIM, stream);
    launch_cvt(wgk1, wgk1b, (size_t)NL * 16 * DIM,  stream);
    launch_cvt(wo,   wob,   (size_t)NL * DIM * DVc, stream);
    launch_cvt(w1,   w1b,   (size_t)NL * FF * DIM,  stream);
    launch_cvt(w2,   w2b,   (size_t)NL * DIM * FF,  stream);
    launch_cvt(head_w, headb, (size_t)VOC * DIM,    stream);
    cvt_pad_wgk2_kernel<<<(NL * DKc * 32 + 255) / 256, 256, 0, stream>>>(wgk2, wgk2b);
    fill0_bf16_kernel<<<(TKN * 32 / 4 + 255) / 256, 256, 0, stream>>>(gtb, TKN * 32 / 4);

    embed_kernel<<<TKN, 256, 0, stream>>>(tokens, embed, x);

    for (int l = 0; l < NL; ++l) {
        const unsigned short* wq_l   = wqb   + (size_t)l * DKc * DIM;
        const unsigned short* wk_l   = wkb   + (size_t)l * DKc * DIM;
        const unsigned short* wv_l   = wvb   + (size_t)l * DVc * DIM;
        const unsigned short* wg_l   = wgb   + (size_t)l * DVc * DIM;
        const unsigned short* wgk1_l = wgk1b + (size_t)l * 16 * DIM;
        const unsigned short* wgk2_l = wgk2b + (size_t)l * DKc * 32;
        const unsigned short* wo_l   = wob   + (size_t)l * DIM * DVc;
        const unsigned short* w1_l   = w1b   + (size_t)l * FF * DIM;
        const unsigned short* w2_l   = w2b   + (size_t)l * DIM * FF;
        const float* bgk_l = bgk + (size_t)l * DKc;
        const float* onw_l = onw + (size_t)l * HVc;
        const float* b1_l  = b1  + (size_t)l * FF;
        const float* b2_l  = b2  + (size_t)l * DIM;

        ln_kernel<<<TKN, 256, 0, stream>>>(x, ln1_w + l * DIM, ln1_b + l * DIM, h);

        launch_gemm(h, wq_l, q, nullptr, DKc, TKN, DKc, DIM, nullptr, nullptr, 0, stream);
        launch_gemm(h, wk_l, k, nullptr, DKc, TKN, DKc, DIM, nullptr, nullptr, 0, stream);
        launch_gemm(h, wv_l, v, nullptr, DVc, TKN, DVc, DIM, nullptr, nullptr, 0, stream);
        launch_gemm(h, wg_l, g, nullptr, DVc, TKN, DVc, DIM, nullptr, nullptr, 0, stream);

        // low-rank gate: (h @ wgk1^T)[T,16] -> bf16 [T,32]; then @ wgk2^T (K=32 padded)
        launch_gemm(h,   wgk1_l, nullptr, gtb, 32,  TKN, 16,  DIM, nullptr, nullptr, 0, stream);
        launch_gemm(gtb, wgk2_l, gk, nullptr, DKc,  TKN, DKc, 32,  bgk_l,   nullptr, 5, stream);

        gla_kernel<<<dim3(NH, BATCH), 128, 0, stream>>>(q, k, v, gk, o);
        gatenorm_kernel<<<dim3(TKN, NH), 64, 0, stream>>>(o, g, onw_l, ob);

        launch_gemm(ob, wo_l, x, nullptr, DIM, TKN, DIM, DVc, nullptr, x, 3, stream); // x += o@wo^T

        ln_kernel<<<TKN, 256, 0, stream>>>(x, ln2_w + l * DIM, ln2_b + l * DIM, h);
        launch_gemm(h,    w1_l, nullptr, midb, FF, TKN, FF,  DIM, b1_l, nullptr, 2, stream); // gelu
        launch_gemm(midb, w2_l, x, nullptr, DIM,   TKN, DIM, FF,  b2_l, x,       4, stream); // +res
    }

    ln_kernel<<<TKN, 256, 0, stream>>>(x, lnf_w, lnf_b, h);
    launch_gemm(h, headb, out, nullptr, VOC, TKN, VOC, DIM, head_b, nullptr, 1, stream);
}
